// RoIHeads_67619965108725
// MI455X (gfx1250) — matrix operand order repaired and checked by reading, not compiled, not run on hardware
//
#include <hip/hip_runtime.h>
#include <hip/hip_bf16.h>
#include <math.h>
#include <stdint.h>

typedef __attribute__((ext_vector_type(16))) __bf16 v16bf;
typedef __attribute__((ext_vector_type(8)))  __bf16 v8bf;
typedef __attribute__((ext_vector_type(8)))  float  v8f;

#define NB          8
#define NPROP       1000
#define NGT         64
#define P_ALL       1064
#define FDIM        12544
#define HDIM        1024
#define NCLS        16
#define NSAMP       512
#define NPOSMAX     128
#define MTOT        (NB * NSAMP)   // 4096
#define SL1_BETA    (1.0f / 9.0f)

// ---------------------------------------------------------------------------
// Kernel 1: IoU matching + deterministic balanced sampling + target encoding
// One block per image. 256 threads.
// ---------------------------------------------------------------------------
__global__ void __launch_bounds__(256)
match_sample_encode(const float* __restrict__ proposals,   // [B,1000,4]
                    const float* __restrict__ gt_bboxes,   // [B,64,4]
                    const float* __restrict__ gt_oboxes,   // [B,64,5]
                    const int*   __restrict__ gt_labels,   // [B,64]
                    int*   __restrict__ rows_g,            // [B*512] gather rows into features
                    int*   __restrict__ labels_out,        // [B*512]
                    float* __restrict__ targets)           // [B*512*5]
{
    __shared__ float s_gt[NGT * 4];
    __shared__ int   s_lab[P_ALL];
    __shared__ int   s_mid[P_ALL];
    __shared__ unsigned char s_keep[P_ALL];
    __shared__ int   s_sidx[NSAMP];

    const int b   = blockIdx.x;
    const int tid = threadIdx.x;

    for (int i = tid; i < NGT * 4; i += blockDim.x)
        s_gt[i] = gt_bboxes[(size_t)b * NGT * 4 + i];
    __syncthreads();

    // --- per-proposal max-IoU match over 64 GTs ---
    for (int p = tid; p < P_ALL; p += blockDim.x) {
        float x0, y0, x1, y1;
        if (p < NPROP) {
            const float* bp = proposals + ((size_t)b * NPROP + p) * 4;
            x0 = bp[0]; y0 = bp[1]; x1 = bp[2]; y1 = bp[3];
        } else {
            const float* bp = s_gt + (p - NPROP) * 4;
            x0 = bp[0]; y0 = bp[1]; x1 = bp[2]; y1 = bp[3];
        }
        const float area_p = (x1 - x0) * (y1 - y0);
        float best = -1.0f;
        int   bi   = 0;
        for (int g = 0; g < NGT; ++g) {
            const float gx0 = s_gt[g * 4 + 0], gy0 = s_gt[g * 4 + 1];
            const float gx1 = s_gt[g * 4 + 2], gy1 = s_gt[g * 4 + 3];
            const float area_g = (gx1 - gx0) * (gy1 - gy0);
            const float w = fmaxf(fminf(gx1, x1) - fmaxf(gx0, x0), 0.0f);
            const float h = fmaxf(fminf(gy1, y1) - fmaxf(gy0, y0), 0.0f);
            const float inter = w * h;
            const float iou = inter / (area_g + area_p - inter);
            if (iou > best) { best = iou; bi = g; }   // strict '>' == first argmax
        }
        int matched = (best < 0.3f) ? -1 : ((best < 0.5f) ? -2 : bi);
        int clamped = matched < 0 ? 0 : matched;
        int lab = (matched == -1) ? 0
                : (matched == -2) ? -1
                : gt_labels[b * NGT + clamped];
        s_lab[p] = lab;
        s_mid[p] = clamped;
    }
    __syncthreads();

    // --- sequential deterministic sampler (matches reference cumsum/top_k) ---
    if (tid == 0) {
        int cum = 0;
        for (int p = 0; p < P_ALL; ++p) {
            const int pos = (s_lab[p] > 0);
            cum += pos;
            s_keep[p] = (pos && cum <= NPOSMAX) ? 1 : 0;
        }
        int npos = 0;
        for (int p = 0; p < P_ALL; ++p) npos += s_keep[p];
        const int nbudget = NSAMP - npos;
        int cn = 0;
        for (int p = 0; p < P_ALL; ++p) {
            if (s_lab[p] == 0) {
                ++cn;
                if (cn <= nbudget) s_keep[p] = 1;
            }
        }
        int cnt = 0;
        for (int p = 0; p < P_ALL && cnt < NSAMP; ++p)
            if (s_keep[p]) s_sidx[cnt++] = p;
        for (int p = 0; p < P_ALL && cnt < NSAMP; ++p)
            if (!s_keep[p]) s_sidx[cnt++] = p;
    }
    __syncthreads();

    // --- gather indices, labels, target encoding ---
    for (int i = tid; i < NSAMP; i += blockDim.x) {
        const int s = s_sidx[i];
        const int m = b * NSAMP + i;
        rows_g[m]     = b * P_ALL + s;
        labels_out[m] = s_lab[s];

        float x0, y0, x1, y1;
        if (s < NPROP) {
            const float* bp = proposals + ((size_t)b * NPROP + s) * 4;
            x0 = bp[0]; y0 = bp[1]; x1 = bp[2]; y1 = bp[3];
        } else {
            const float* bp = s_gt + (s - NPROP) * 4;
            x0 = bp[0]; y0 = bp[1]; x1 = bp[2]; y1 = bp[3];
        }
        const int g = s_mid[s];
        const float* ob = gt_oboxes + ((size_t)b * NGT + g) * 5;
        const float pw  = x1 - x0;
        const float ph  = y1 - y0;
        const float pcx = x0 + 0.5f * pw;
        const float pcy = y0 + 0.5f * ph;
        float* t = targets + (size_t)m * 5;
        t[0] = 10.0f * (ob[0] - pcx) / pw;
        t[1] = 10.0f * (ob[1] - pcy) / ph;
        t[2] = 5.0f  * logf(ob[2] / pw);
        t[3] = 5.0f  * logf(ob[3] / ph);
        t[4] = 10.0f * ob[4];
    }
}

// ---------------------------------------------------------------------------
// Kernel 2: Y = relu(X @ W + bias), fp32 in/out, bf16 WMMA compute.
// Block = 128 rows x 64 cols, 8 waves of 32; each wave owns a 16x64 strip.
// K-step 64, double-buffered LDS B-staging (one barrier per step), all 8 B
// fragments preloaded so the 8 v_wmma issue back-to-back.
// `rows` (nullable) gathers X rows.  K % 64 == 0, N % 64 == 0, M % 128 == 0.
// ---------------------------------------------------------------------------
__global__ void __launch_bounds__(256)
wmma_gemm_relu(const float* __restrict__ X,
               const int*   __restrict__ rows,     // nullable gather
               const float* __restrict__ W,        // [K,N] row-major
               const float* __restrict__ bias,     // [N]
               float* __restrict__ Y,              // [M,N]
               int K, int N)
{
    // frag-ready bf16 staging of a 64(K) x 64(N) B tile, double buffered:
    // slot(fi = ks*4+j, lane, e) holds
    //   B[k = ks*32 + (lane>=16?16:0) + e][n = j*16 + (lane&15)]
    __shared__ __align__(32) __bf16 sB[2][8 * 32 * 16];   // 2 x 8 KB

    const int tid  = threadIdx.x;
    const int wave = tid >> 5;
    const int lane = tid & 31;
    const int half = lane >> 4;
    const int col  = lane & 15;

    const int mbase = blockIdx.x * 128 + wave * 16;
    const int nbase = blockIdx.y * 64;

    // per-lane A row pointer (row = mbase + (lane&15), optionally gathered)
    const int m_a  = mbase + col;
    const int row  = rows ? rows[m_a] : m_a;
    const float* xr = X + (size_t)row * K;

    // staging geometry (thread-invariant): thread == slot id
    const int fi   = tid >> 5;               // 0..7  (= ks*4 + j)
    const int ksS  = fi >> 2;                // 0..1
    const int jS   = fi & 3;                 // 0..3
    const int lS   = tid & 31;
    const int nS   = nbase + jS * 16 + (lS & 15);
    const int kOff = ksS * 32 + ((lS & 16) ? 16 : 0);
    __bf16* mySlot = &sB[0][tid * 16];       // buffer 0 slot; buffer 1 = +8*32*16

    // ---- prologue: stage first tile into buffer 0 ----
    {
        const float* wp = W + (size_t)kOff * N + nS;
        v8bf t0, t1;
#pragma unroll
        for (int u = 0; u < 8; ++u) t0[u] = (__bf16)wp[(size_t)u * N];
#pragma unroll
        for (int u = 0; u < 8; ++u) t1[u] = (__bf16)wp[(size_t)(u + 8) * N];
        *reinterpret_cast<v8bf*>(mySlot)     = t0;
        *reinterpret_cast<v8bf*>(mySlot + 8) = t1;
    }
    __syncthreads();

    v8f acc[4] = {};
    int buf = 0;

    for (int k0 = 0; k0 < K; k0 += 64) {
        const bool more = (k0 + 64) < K;

        // ---- issue staging global loads for the NEXT tile (latency hidden
        //      under this tile's WMMAs) ----
        float wnext[16];
        if (more) {
            const float* wp = W + (size_t)(k0 + 64 + kOff) * N + nS;
#pragma unroll
            for (int u = 0; u < 16; ++u) wnext[u] = wp[(size_t)u * N];
        }

        // ---- A fragments (direct global, gathered row) ----
        // lane<16 -> K {0..15}, lane>=16 -> {16..31} within each 32
        v16bf a0, a1;
#pragma unroll
        for (int ks = 0; ks < 2; ++ks) {
            const float* p0 = xr + k0 + ks * 32 + (half << 4);
            const float4 f0 = *reinterpret_cast<const float4*>(p0);
            const float4 f1 = *reinterpret_cast<const float4*>(p0 + 4);
            const float4 g0 = *reinterpret_cast<const float4*>(p0 + 8);
            const float4 g1 = *reinterpret_cast<const float4*>(p0 + 12);
            v16bf av;
            av[0]  = (__bf16)f0.x; av[1]  = (__bf16)f0.y; av[2]  = (__bf16)f0.z; av[3]  = (__bf16)f0.w;
            av[4]  = (__bf16)f1.x; av[5]  = (__bf16)f1.y; av[6]  = (__bf16)f1.z; av[7]  = (__bf16)f1.w;
            av[8]  = (__bf16)g0.x; av[9]  = (__bf16)g0.y; av[10] = (__bf16)g0.z; av[11] = (__bf16)g0.w;
            av[12] = (__bf16)g1.x; av[13] = (__bf16)g1.y; av[14] = (__bf16)g1.z; av[15] = (__bf16)g1.w;
            if (ks == 0) a0 = av; else a1 = av;
        }
        if (more)
            __builtin_prefetch(xr + k0 + 64 + (half << 3), 0, 3); // global_prefetch

        // ---- preload all 8 B fragments, then 8 back-to-back WMMAs ----
        const __bf16* sb = &sB[buf][0];
        v16bf bf[8];
#pragma unroll
        for (int f = 0; f < 8; ++f)
            bf[f] = *reinterpret_cast<const v16bf*>(&sb[(f * 32 + lane) * 16]);

#pragma unroll
        for (int j = 0; j < 4; ++j)
            acc[j] = __builtin_amdgcn_wmma_f32_16x16x32_bf16(
                false, a0, false, bf[j], (short)0, acc[j], false, false);
#pragma unroll
        for (int j = 0; j < 4; ++j)
            acc[j] = __builtin_amdgcn_wmma_f32_16x16x32_bf16(
                false, a1, false, bf[4 + j], (short)0, acc[j], false, false);

        // ---- convert + store NEXT tile into the other buffer ----
        if (more) {
            __bf16* dst = mySlot + (buf ^ 1) * (8 * 32 * 16);
            v8bf t0, t1;
#pragma unroll
            for (int u = 0; u < 8; ++u) t0[u] = (__bf16)wnext[u];
#pragma unroll
            for (int u = 0; u < 8; ++u) t1[u] = (__bf16)wnext[u + 8];
            *reinterpret_cast<v8bf*>(dst)     = t0;
            *reinterpret_cast<v8bf*>(dst + 8) = t1;
        }
        __syncthreads();
        buf ^= 1;
    }

    // ---- epilogue: bias + relu; C layout: vgpr r -> m = r + 8*half, n = col
#pragma unroll
    for (int j = 0; j < 4; ++j) {
        const int n  = nbase + j * 16 + col;
        const float bv = bias[n];
#pragma unroll
        for (int r = 0; r < 8; ++r) {
            const int m = mbase + r + half * 8;
            float v = acc[j][r] + bv;
            Y[(size_t)m * N + n] = v > 0.0f ? v : 0.0f;
        }
    }
}

// ---------------------------------------------------------------------------
// Kernel 3: cls logits + per-class reg columns + CE / smooth-L1 partial sums.
// One thread per sample; 16 blocks x 256 threads.
// ---------------------------------------------------------------------------
__global__ void __launch_bounds__(256)
head_loss(const float* __restrict__ H2,       // [M,1024]
          const float* __restrict__ cls_w,    // [1024,16]
          const float* __restrict__ cls_b,    // [16]
          const float* __restrict__ reg_w,    // [1024,80]
          const float* __restrict__ reg_b,    // [80]
          const int*   __restrict__ labels,   // [M]
          const float* __restrict__ targets,  // [M,5]
          float* __restrict__ partial)        // [2*gridDim.x]
{
    __shared__ float red[256];
    const int tid = threadIdx.x;
    const int m   = blockIdx.x * blockDim.x + tid;

    const float* hr = H2 + (size_t)m * HDIM;
    const int lab = labels[m];
    const int lc  = (lab < 0) ? lab + NCLS : lab;   // python-style negative wrap

    float lg[NCLS];
#pragma unroll
    for (int c = 0; c < NCLS; ++c) lg[c] = 0.0f;
    float rg[5] = {0, 0, 0, 0, 0};

    for (int k = 0; k < HDIM; ++k) {
        const float hv = hr[k];
        const float4* cw = reinterpret_cast<const float4*>(cls_w + (size_t)k * NCLS);
        const float4 c0 = cw[0], c1 = cw[1], c2 = cw[2], c3 = cw[3];
        lg[0]  += hv * c0.x; lg[1]  += hv * c0.y; lg[2]  += hv * c0.z; lg[3]  += hv * c0.w;
        lg[4]  += hv * c1.x; lg[5]  += hv * c1.y; lg[6]  += hv * c1.z; lg[7]  += hv * c1.w;
        lg[8]  += hv * c2.x; lg[9]  += hv * c2.y; lg[10] += hv * c2.z; lg[11] += hv * c2.w;
        lg[12] += hv * c3.x; lg[13] += hv * c3.y; lg[14] += hv * c3.z; lg[15] += hv * c3.w;
        const float* rw = reg_w + (size_t)k * (NCLS * 5) + lc * 5;
#pragma unroll
        for (int j = 0; j < 5; ++j) rg[j] += hv * rw[j];
    }

    float maxl = -1e30f;
#pragma unroll
    for (int c = 0; c < NCLS; ++c) { lg[c] += cls_b[c]; maxl = fmaxf(maxl, lg[c]); }
    float se = 0.0f;
#pragma unroll
    for (int c = 0; c < NCLS; ++c) se += expf(lg[c] - maxl);
    const float lse  = logf(se) + maxl;
    const float lcls = lse - lg[lc];

    float lreg = 0.0f;
    if (lab > 0) {
        const float* t = targets + (size_t)m * 5;
#pragma unroll
        for (int j = 0; j < 5; ++j) {
            const float d = fabsf(rg[j] + reg_b[lc * 5 + j] - t[j]);
            lreg += (d < SL1_BETA) ? (0.5f * d * d / SL1_BETA) : (d - 0.5f * SL1_BETA);
        }
    }

    // deterministic tree reductions
    red[tid] = lcls; __syncthreads();
    for (int s = 128; s > 0; s >>= 1) {
        if (tid < s) red[tid] += red[tid + s];
        __syncthreads();
    }
    if (tid == 0) partial[blockIdx.x] = red[0];
    __syncthreads();

    red[tid] = lreg; __syncthreads();
    for (int s = 128; s > 0; s >>= 1) {
        if (tid < s) red[tid] += red[tid + s];
        __syncthreads();
    }
    if (tid == 0) partial[gridDim.x + blockIdx.x] = red[0];
}

__global__ void finalize(const float* __restrict__ partial, float* __restrict__ out,
                         int nblk)
{
    if (blockIdx.x == 0 && threadIdx.x == 0) {
        float c = 0.0f, r = 0.0f;
        for (int i = 0; i < nblk; ++i) { c += partial[i]; r += partial[nblk + i]; }
        out[0] = c / (float)MTOT;
        out[1] = r / (float)MTOT;
    }
}

// ---------------------------------------------------------------------------
extern "C" void kernel_launch(void* const* d_in, const int* in_sizes, int n_in,
                              void* d_out, int out_size, void* d_ws, size_t ws_size,
                              hipStream_t stream) {
    const float* proposals = (const float*)d_in[0];
    const float* gt_bboxes = (const float*)d_in[1];
    const float* gt_oboxes = (const float*)d_in[2];
    const int*   gt_labels = (const int*)  d_in[3];
    const float* features  = (const float*)d_in[4];
    const float* fc1_w     = (const float*)d_in[5];
    const float* fc1_b     = (const float*)d_in[6];
    const float* fc2_w     = (const float*)d_in[7];
    const float* fc2_b     = (const float*)d_in[8];
    const float* cls_w     = (const float*)d_in[9];
    const float* cls_b     = (const float*)d_in[10];
    const float* reg_w     = (const float*)d_in[11];
    const float* reg_b     = (const float*)d_in[12];
    float* out = (float*)d_out;

    uintptr_t p = (uintptr_t)d_ws;
    auto take = [&](size_t bytes) -> void* {
        p = (p + 255) & ~(uintptr_t)255;
        void* r = (void*)p;
        p += bytes;
        return r;
    };
    int*   rows_g  = (int*)  take((size_t)MTOT * sizeof(int));
    int*   labels  = (int*)  take((size_t)MTOT * sizeof(int));
    float* targets = (float*)take((size_t)MTOT * 5 * sizeof(float));
    float* H1      = (float*)take((size_t)MTOT * HDIM * sizeof(float));
    float* H2      = (float*)take((size_t)MTOT * HDIM * sizeof(float));
    float* partial = (float*)take(2 * 16 * sizeof(float));

    match_sample_encode<<<NB, 256, 0, stream>>>(
        proposals, gt_bboxes, gt_oboxes, gt_labels, rows_g, labels, targets);

    // H1 = relu(gather(features, rows) @ fc1_w + fc1_b)   M=4096 K=12544 N=1024
    wmma_gemm_relu<<<dim3(MTOT / 128, HDIM / 64), 256, 0, stream>>>(
        features, rows_g, fc1_w, fc1_b, H1, FDIM, HDIM);

    // H2 = relu(H1 @ fc2_w + fc2_b)                        M=4096 K=1024  N=1024
    wmma_gemm_relu<<<dim3(MTOT / 128, HDIM / 64), 256, 0, stream>>>(
        H1, nullptr, fc2_w, fc2_b, H2, HDIM, HDIM);

    head_loss<<<MTOT / 256, 256, 0, stream>>>(
        H2, cls_w, cls_b, reg_w, reg_b, labels, targets, partial);

    finalize<<<1, 32, 0, stream>>>(partial, out, MTOT / 256);
}
